// CausalSelfAttention_15470472200618
// MI455X (gfx1250) — compile-verified, matrix operand order checked by reading, not run on hardware
//
#include <hip/hip_runtime.h>

// ---------------------------------------------------------------------------
// CDNA5 / gfx1250 causal self-attention (GPT-2 small shapes)
//   B=2, T=4096, C=768, H=12, hd=64.
//   All matmuls via v_wmma_f32_16x16x32_bf16; LDS staging uses
//   global_load_async_to_lds_b128 (ASYNCcnt) where the copy is conversion-free.
// ---------------------------------------------------------------------------

typedef __attribute__((ext_vector_type(16))) __bf16 v16bf;
typedef __attribute__((ext_vector_type(8)))  float  v8f;

union FragBF {
    unsigned int u[8];
    v16bf v;
};

__device__ __forceinline__ unsigned short f2bf(float f) {
    unsigned int u = __float_as_uint(f);
    unsigned int r = (u + 0x7FFFu + ((u >> 16) & 1u)) >> 16;  // RNE
    return (unsigned short)r;
}
__device__ __forceinline__ unsigned int pack2bf(float a, float b) {
    return (unsigned int)f2bf(a) | ((unsigned int)f2bf(b) << 16);
}

// -------- async global->LDS (CDNA5 ASYNCcnt path), guarded ----------------
#if defined(__has_builtin)
#  if __has_builtin(__builtin_amdgcn_global_load_async_to_lds_b128)
#    define HAVE_ASYNC_LDS 1
#  endif
#endif
#ifndef HAVE_ASYNC_LDS
#  define HAVE_ASYNC_LDS 0
#endif

#if HAVE_ASYNC_LDS
typedef int v4i_vs __attribute__((vector_size(16)));   // int4 vector, gcc-style

__device__ __forceinline__ void async_b128(const void* g, void* l) {
    __builtin_amdgcn_global_load_async_to_lds_b128(
        (__attribute__((address_space(1))) v4i_vs*)(unsigned long long)(g),
        (__attribute__((address_space(3))) v4i_vs*)(l), 0, 0);
}
__device__ __forceinline__ void wait_async0() {
#  if __has_builtin(__builtin_amdgcn_s_wait_asynccnt)
    __builtin_amdgcn_s_wait_asynccnt(0);
#  else
    asm volatile("s_wait_asynccnt 0x0" ::: "memory");
#  endif
}
#endif

constexpr int Bsz  = 2;
constexpr int Tlen = 4096;
constexpr int C    = 768;
constexpr int NH   = 12;
constexpr int HD   = 64;
constexpr int Mtot = Bsz * Tlen;   // 8192
constexpr int C3   = 3 * C;        // 2304
constexpr int AS   = 40;           // GEMM LDS stride: 80B rows (20 banks, conflict-free, 16B aligned)
constexpr int KS   = 72;           // attn LDS stride: 144B rows (36 banks, conflict-free, 16B aligned)

// ===========================================================================
// Kernel 1: QKV projection.  x[Mtot,C] f32 @ w[C,3C] f32 + b  ->  q/k/v bf16
//           in head-major layout [Bsz*NH, Tlen, HD].
// Block: 256 thr (8 waves, 2x4), tile 128x128, K-step 32.
// ===========================================================================
__global__ __launch_bounds__(256)
void qkv_gemm(const float* __restrict__ x, const float* __restrict__ w,
              const float* __restrict__ bias,
              unsigned short* __restrict__ qout, unsigned short* __restrict__ kout,
              unsigned short* __restrict__ vout) {
    __shared__ unsigned short ldsA[128 * AS];   // [m][k] bf16
    __shared__ unsigned short ldsB[128 * AS];   // [n][k] bf16 (transposed tile)
    const int m0 = blockIdx.x * 128;
    const int n0 = blockIdx.y * 128;
    const int tid  = threadIdx.x;
    const int lane = tid & 31, wid = tid >> 5;
    const int wm = wid >> 2, wn = wid & 3;
    const int lm = lane & 15, lh = lane >> 4;

    v8f zero = {};
    v8f acc[4][2];
    for (int i = 0; i < 4; i++)
        for (int j = 0; j < 2; j++) acc[i][j] = zero;

    for (int k0 = 0; k0 < C; k0 += 32) {
        // A tile 128x32: fp32 loads -> bf16 convert -> LDS
        for (int it = 0; it < 4; it++) {
            int f = tid + it * 256;
            int row = f >> 3, kq = (f & 7) * 4;
            const float4 xv = *(const float4*)(x + (size_t)(m0 + row) * C + k0 + kq);
            *(uint2*)&ldsA[row * AS + kq] =
                make_uint2(pack2bf(xv.x, xv.y), pack2bf(xv.z, xv.w));
        }
        // B tile 32x128, transposed into ldsB[n][k]
        for (int it = 0; it < 4; it++) {
            int f = tid + it * 256;
            int kr = f >> 5, nq = (f & 31) * 4;
            const float4 wv = *(const float4*)(w + (size_t)(k0 + kr) * C3 + n0 + nq);
            ldsB[(nq + 0) * AS + kr] = f2bf(wv.x);
            ldsB[(nq + 1) * AS + kr] = f2bf(wv.y);
            ldsB[(nq + 2) * AS + kr] = f2bf(wv.z);
            ldsB[(nq + 3) * AS + kr] = f2bf(wv.w);
        }
        __syncthreads();

        FragBF bf[2];
        for (int j = 0; j < 2; j++) {
            int n = wn * 32 + j * 16 + lm;
            for (int v = 0; v < 8; v++) {
                int kk = ((v >> 2) << 4) + lh * 8 + ((v & 3) << 1);
                bf[j].u[v] = *(const unsigned int*)&ldsB[n * AS + kk];
            }
        }
        for (int i = 0; i < 4; i++) {
            FragBF af;
            int m = wm * 64 + i * 16 + lm;
            for (int v = 0; v < 8; v++) {
                int kk = ((v >> 2) << 4) + lh * 8 + ((v & 3) << 1);
                af.u[v] = *(const unsigned int*)&ldsA[m * AS + kk];
            }
            for (int j = 0; j < 2; j++)
                acc[i][j] = __builtin_amdgcn_wmma_f32_16x16x32_bf16(
                    false, af.v, false, bf[j].v, (short)0, acc[i][j], false, false);
        }
        __syncthreads();
    }

    // epilogue: bias + scatter into head-major q/k/v (bf16)
    for (int j = 0; j < 2; j++) {
        int n = n0 + wn * 32 + j * 16 + lm;
        float bv = bias[n];
        int which = n / C, cj = n % C;
        int h = cj / HD, d = cj % HD;
        unsigned short* dst = (which == 0) ? qout : (which == 1 ? kout : vout);
        for (int i = 0; i < 4; i++) {
            int mbase = m0 + wm * 64 + i * 16;
            for (int v = 0; v < 8; v++) {
                int m = mbase + v + lh * 8;
                int b = m / Tlen, t = m % Tlen;
                dst[((size_t)(b * NH + h) * Tlen + t) * HD + d] = f2bf(acc[i][j][v] + bv);
            }
        }
    }
}

// ===========================================================================
// Kernel 2: flash attention (causal, online softmax).
// Block: 128 thr (4 waves); block = 64 query rows of one (b,h); wave = 16 rows.
// K tile streamed via global_load_async_to_lds_b128; V transposed by VALU.
// ===========================================================================
__global__ __launch_bounds__(128)
void flash_attn(const unsigned short* __restrict__ qb,
                const unsigned short* __restrict__ kb,
                const unsigned short* __restrict__ vb,
                unsigned short* __restrict__ yout) {
    __shared__ unsigned short ldsK[64 * KS];      // [key][d]
    __shared__ unsigned short ldsV[64 * KS];      // transposed: [d][key]
    __shared__ unsigned short ldsP[4][16 * KS];   // per-wave P tile [m][key]

    const int bh = blockIdx.y;             // b*NH + h
    const int q0 = blockIdx.x * 64;
    const int tid = threadIdx.x, lane = tid & 31, wid = tid >> 5;
    const int lm = lane & 15, lh = lane >> 4;
    const size_t head_base = (size_t)bh * Tlen * HD;

    // Q fragments (A-layout, hd=64 -> two K=32 fragments), loaded once.
    FragBF qf[2];
    {
        const unsigned short* qrow = qb + head_base + (size_t)(q0 + wid * 16 + lm) * HD;
        for (int half = 0; half < 2; half++)
            for (int v = 0; v < 8; v++) {
                int d = half * 32 + ((v >> 2) << 4) + lh * 8 + ((v & 3) << 1);
                qf[half].u[v] = *(const unsigned int*)(qrow + d);
            }
    }

    v8f zero = {};
    v8f o[4];
    for (int dt = 0; dt < 4; dt++) o[dt] = zero;
    float mrow[8], lrow[8];
    for (int r = 0; r < 8; r++) { mrow[r] = -1e30f; lrow[r] = 0.0f; }

    const float scale = 0.125f;           // 1/sqrt(64)
    const int nkb = blockIdx.x + 1;       // causal: keys up to end of query tile

    for (int kbi = 0; kbi < nkb; kbi++) {
        __syncthreads();                  // protect LDS reuse from previous iter
        const unsigned short* ktile = kb + head_base + (size_t)kbi * 64 * HD;
        const unsigned short* vtile = vb + head_base + (size_t)kbi * 64 * HD;
        if (kbi + 1 < nkb) {              // prefetch next tiles (global_prefetch_b8)
            __builtin_prefetch(ktile + 64 * HD + tid * 64, 0, 1);
            __builtin_prefetch(vtile + 64 * HD + tid * 64, 0, 1);
        }
        // K tile: straight copy -> async DMA to LDS; V tile: transpose via VALU.
        for (int it = 0; it < 4; it++) {
            int f = tid + it * 128;       // uint4 units, [0,512)
            int row = f >> 3, dq = (f & 7) * 8;
#if HAVE_ASYNC_LDS
            async_b128(ktile + (size_t)row * HD + dq, &ldsK[row * KS + dq]);
#else
            uint4 u = *(const uint4*)(ktile + (size_t)row * HD + dq);
            *(uint2*)&ldsK[row * KS + dq]     = make_uint2(u.x, u.y);
            *(uint2*)&ldsK[row * KS + dq + 4] = make_uint2(u.z, u.w);
#endif
            uint4 vv = *(const uint4*)(vtile + (size_t)row * HD + dq);
            unsigned int wv[4] = {vv.x, vv.y, vv.z, vv.w};
            for (int i2 = 0; i2 < 4; i2++) {
                ldsV[(dq + 2 * i2 + 0) * KS + row] = (unsigned short)(wv[i2] & 0xffffu);
                ldsV[(dq + 2 * i2 + 1) * KS + row] = (unsigned short)(wv[i2] >> 16);
            }
        }
#if HAVE_ASYNC_LDS
        wait_async0();
#endif
        __syncthreads();

        // S = Q K^T : 4 key-subtiles x (2 chained K=32 WMMAs)
        v8f s[4];
        for (int jt = 0; jt < 4; jt++) {
            FragBF b0, b1;
            int key = jt * 16 + lm;
            for (int v = 0; v < 8; v++) {
                int kk = ((v >> 2) << 4) + lh * 8 + ((v & 3) << 1);
                b0.u[v] = *(const unsigned int*)&ldsK[key * KS + kk];
                b1.u[v] = *(const unsigned int*)&ldsK[key * KS + kk + 32];
            }
            v8f t0 = __builtin_amdgcn_wmma_f32_16x16x32_bf16(
                false, qf[0].v, false, b0.v, (short)0, zero, false, false);
            s[jt] = __builtin_amdgcn_wmma_f32_16x16x32_bf16(
                false, qf[1].v, false, b1.v, (short)0, t0, false, false);
        }
        // scale + causal mask (NaN-safe finite mask value)
        for (int jt = 0; jt < 4; jt++) {
            int keyg = kbi * 64 + jt * 16 + lm;
            for (int r = 0; r < 8; r++) {
                int qg = q0 + wid * 16 + r + lh * 8;
                float sv = s[jt][r] * scale;
                s[jt][r] = (keyg > qg) ? -1e30f : sv;
            }
        }
        // online softmax: row stats via half-wave butterfly reductions
        float pr[4][8];
        for (int r = 0; r < 8; r++) {
            float v0 = fmaxf(fmaxf(s[0][r], s[1][r]), fmaxf(s[2][r], s[3][r]));
            v0 = fmaxf(v0, __shfl_xor(v0, 1, 32));
            v0 = fmaxf(v0, __shfl_xor(v0, 2, 32));
            v0 = fmaxf(v0, __shfl_xor(v0, 4, 32));
            v0 = fmaxf(v0, __shfl_xor(v0, 8, 32));
            float mnew  = fmaxf(mrow[r], v0);
            float alpha = __expf(mrow[r] - mnew);
            mrow[r] = mnew;
            float rs = 0.0f;
            for (int jt = 0; jt < 4; jt++) {
                float p = __expf(s[jt][r] - mnew);
                pr[jt][r] = p;
                rs += p;
            }
            rs += __shfl_xor(rs, 1, 32);
            rs += __shfl_xor(rs, 2, 32);
            rs += __shfl_xor(rs, 4, 32);
            rs += __shfl_xor(rs, 8, 32);
            lrow[r] = lrow[r] * alpha + rs;
            for (int dt = 0; dt < 4; dt++) o[dt][r] *= alpha;
        }
        // C-layout -> A-layout via per-wave LDS round trip
        for (int jt = 0; jt < 4; jt++) {
            int key = jt * 16 + lm;
            for (int r = 0; r < 8; r++)
                ldsP[wid][(r + lh * 8) * KS + key] = f2bf(pr[jt][r]);
        }
        __syncthreads();
        FragBF pf[2];
        for (int half = 0; half < 2; half++)
            for (int v = 0; v < 8; v++) {
                int kk = half * 32 + ((v >> 2) << 4) + lh * 8 + ((v & 3) << 1);
                pf[half].u[v] = *(const unsigned int*)&ldsP[wid][lm * KS + kk];
            }
        // O += P V
        for (int dt = 0; dt < 4; dt++) {
            FragBF vb0, vb1;
            int dcol = dt * 16 + lm;
            for (int v = 0; v < 8; v++) {
                int kk = ((v >> 2) << 4) + lh * 8 + ((v & 3) << 1);
                vb0.u[v] = *(const unsigned int*)&ldsV[dcol * KS + kk];
                vb1.u[v] = *(const unsigned int*)&ldsV[dcol * KS + kk + 32];
            }
            o[dt] = __builtin_amdgcn_wmma_f32_16x16x32_bf16(
                false, pf[0].v, false, vb0.v, (short)0, o[dt], false, false);
            o[dt] = __builtin_amdgcn_wmma_f32_16x16x32_bf16(
                false, pf[1].v, false, vb1.v, (short)0, o[dt], false, false);
        }
    }

    // normalize + write att output as [B*T, C] bf16 (ready for out-proj GEMM)
    const int b = bh / NH, h = bh % NH;
    for (int dt = 0; dt < 4; dt++)
        for (int r = 0; r < 8; r++) {
            int q = q0 + wid * 16 + r + lh * 8;
            float val = o[dt][r] / lrow[r];
            yout[((size_t)(b * Tlen + q)) * C + h * HD + dt * 16 + lm] = f2bf(val);
        }
}

// ===========================================================================
// Kernel 3: output projection.  att[Mtot,C] bf16 @ w_proj[C,C] f32 + b -> f32
// A tile is pure bf16 copy -> async DMA to LDS.
// ===========================================================================
__global__ __launch_bounds__(256)
void out_gemm(const unsigned short* __restrict__ a, const float* __restrict__ w,
              const float* __restrict__ bias, float* __restrict__ out) {
    __shared__ unsigned short ldsA[128 * AS];
    __shared__ unsigned short ldsB[128 * AS];
    const int m0 = blockIdx.x * 128;
    const int n0 = blockIdx.y * 128;
    const int tid = threadIdx.x, lane = tid & 31, wid = tid >> 5;
    const int wm = wid >> 2, wn = wid & 3;
    const int lm = lane & 15, lh = lane >> 4;

    v8f zero = {};
    v8f acc[4][2];
    for (int i = 0; i < 4; i++)
        for (int j = 0; j < 2; j++) acc[i][j] = zero;

    for (int k0 = 0; k0 < C; k0 += 32) {
        // A tile already bf16: async copy 512 x b128
        for (int it = 0; it < 2; it++) {
            int f = tid + it * 256;
            int row = f >> 2, kq = (f & 3) * 8;
#if HAVE_ASYNC_LDS
            async_b128(a + (size_t)(m0 + row) * C + k0 + kq, &ldsA[row * AS + kq]);
#else
            uint4 u = *(const uint4*)(a + (size_t)(m0 + row) * C + k0 + kq);
            *(uint2*)&ldsA[row * AS + kq]     = make_uint2(u.x, u.y);
            *(uint2*)&ldsA[row * AS + kq + 4] = make_uint2(u.z, u.w);
#endif
        }
        // B tile 32x128 f32, transposed into ldsB[n][k]
        for (int it = 0; it < 4; it++) {
            int f = tid + it * 256;
            int kr = f >> 5, nq = (f & 31) * 4;
            const float4 wv = *(const float4*)(w + (size_t)(k0 + kr) * C + n0 + nq);
            ldsB[(nq + 0) * AS + kr] = f2bf(wv.x);
            ldsB[(nq + 1) * AS + kr] = f2bf(wv.y);
            ldsB[(nq + 2) * AS + kr] = f2bf(wv.z);
            ldsB[(nq + 3) * AS + kr] = f2bf(wv.w);
        }
#if HAVE_ASYNC_LDS
        wait_async0();
#endif
        __syncthreads();

        FragBF bf[2];
        for (int j = 0; j < 2; j++) {
            int n = wn * 32 + j * 16 + lm;
            for (int v = 0; v < 8; v++) {
                int kk = ((v >> 2) << 4) + lh * 8 + ((v & 3) << 1);
                bf[j].u[v] = *(const unsigned int*)&ldsB[n * AS + kk];
            }
        }
        for (int i = 0; i < 4; i++) {
            FragBF af;
            int m = wm * 64 + i * 16 + lm;
            for (int v = 0; v < 8; v++) {
                int kk = ((v >> 2) << 4) + lh * 8 + ((v & 3) << 1);
                af.u[v] = *(const unsigned int*)&ldsA[m * AS + kk];
            }
            for (int j = 0; j < 2; j++)
                acc[i][j] = __builtin_amdgcn_wmma_f32_16x16x32_bf16(
                    false, af.v, false, bf[j].v, (short)0, acc[i][j], false, false);
        }
        __syncthreads();
    }

    for (int j = 0; j < 2; j++) {
        int n = n0 + wn * 32 + j * 16 + lm;
        float bv = bias[n];
        for (int i = 0; i < 4; i++) {
            int mbase = m0 + wm * 64 + i * 16;
            for (int v = 0; v < 8; v++) {
                int m = mbase + v + lh * 8;
                out[(size_t)m * C + n] = acc[i][j][v] + bv;
            }
        }
    }
}

// ===========================================================================
extern "C" void kernel_launch(void* const* d_in, const int* in_sizes, int n_in,
                              void* d_out, int out_size, void* d_ws, size_t ws_size,
                              hipStream_t stream) {
    const float* x      = (const float*)d_in[0];
    const float* w_attn = (const float*)d_in[1];
    const float* b_attn = (const float*)d_in[2];
    const float* w_proj = (const float*)d_in[3];
    const float* b_proj = (const float*)d_in[4];
    float* out = (float*)d_out;

    const size_t plane = (size_t)Mtot * C;         // 6.29M bf16 elems per buffer
    unsigned short* qws = (unsigned short*)d_ws;
    unsigned short* kws = qws + plane;
    unsigned short* vws = kws + plane;
    unsigned short* yws = vws + plane;

    dim3 g1(Mtot / 128, C3 / 128);                 // 64 x 18
    qkv_gemm<<<g1, 256, 0, stream>>>(x, w_attn, b_attn, qws, kws, vws);

    dim3 g2(Tlen / 64, Bsz * NH);                  // 64 x 24
    flash_attn<<<g2, 128, 0, stream>>>(qws, kws, vws, yws);

    dim3 g3(Mtot / 128, C / 128);                  // 64 x 6
    out_gemm<<<g3, 256, 0, stream>>>(yws, w_proj, b_proj, out);
}
